// GNN_62079457296459
// MI455X (gfx1250) — matrix operand order repaired and checked by reading, not compiled, not run on hardware
//
#include <hip/hip_runtime.h>

typedef __attribute__((ext_vector_type(16))) __bf16 v16bf;
typedef __attribute__((ext_vector_type(8)))  __bf16 v8bf;
typedef __attribute__((ext_vector_type(8)))  float  v8f;

#define DEVINL __device__ __forceinline__

DEVINL void split_bf16(float x, __bf16& hi, __bf16& lo) {
    hi = (__bf16)x;
    lo = (__bf16)(x - (float)hi);
}

DEVINL v16bf cat8(v8bf a, v8bf b) {
    return __builtin_shufflevector(a, b, 0,1,2,3,4,5,6,7,8,9,10,11,12,13,14,15);
}

DEVINL v16bf ldA(const __bf16* __restrict__ p) {        // 16 bf16: [k0..k0+7 | k0+16..k0+23]
    return cat8(*(const v8bf*)(p), *(const v8bf*)(p + 16));
}
DEVINL v16bf ldB(const __bf16* __restrict__ p) {        // 16 bf16: [k0..k0+7 | k0+16..k0+23]
    return cat8(*(const v8bf*)(p), *(const v8bf*)(p + 16));
}

DEVINL v8f wmma_bf16(v16bf a, v16bf b, v8f c) {
    return __builtin_amdgcn_wmma_f32_16x16x32_bf16(false, a, false, b, (short)0, c, false, false);
}

// ---------------- fp32 -> (hi, lo) bf16 split ----------------
__global__ void gnn_cvt_split(const float* __restrict__ in,
                              __bf16* __restrict__ oh, __bf16* __restrict__ ol, int n) {
    int i = blockIdx.x * 256 + threadIdx.x;
    if (i < n) {
        __bf16 h, l;
        split_bf16(in[i], h, l);
        oh[i] = h; ol[i] = l;
    }
}

// ---- weight prep: wcatT[l][n][k] = (k<256 ? w_rel[l][k][n] : w_root[l][k-256][n]) ----
__global__ void gnn_prep_wcat(const float* __restrict__ w_rel, const float* __restrict__ w_root,
                              __bf16* __restrict__ wh, __bf16* __restrict__ wl) {
    int i = blockIdx.x * 256 + threadIdx.x;      // 1536 blocks x 256 == 3*256*512 exactly
    int l = i >> 17;
    int r = i & 131071;
    int n = r >> 9;
    int k = r & 511;
    float v = (k < 256) ? w_rel[l * 65536 + k * 256 + n]
                        : w_root[l * 65536 + (k - 256) * 256 + n];
    __bf16 h, lo; split_bf16(v, h, lo);
    wh[i] = h; wl[i] = lo;
}

// ---- woutT[n][k] = w_out[k][n], n<128, k<256 (128 blocks x 256) ----
__global__ void gnn_prep_wout(const float* __restrict__ w_out,
                              __bf16* __restrict__ wh, __bf16* __restrict__ wl) {
    int i = blockIdx.x * 256 + threadIdx.x;
    int n = i >> 8;
    int k = i & 255;
    float v = w_out[k * 128 + n];
    __bf16 h, lo; split_bf16(v, h, lo);
    wh[i] = h; wl[i] = lo;
}

// ---------------- gather + scale + scatter-add (memory-roofline part) ----------------
__global__ void gnn_scatter(const int* __restrict__ ei, const float* __restrict__ ea,
                            const __bf16* __restrict__ hH, const __bf16* __restrict__ hL,
                            float* __restrict__ agg, int E) {
    int e = blockIdx.x;
    int t = threadIdx.x;
    int s = ei[e];        // source node j
    int d = ei[E + e];    // target node i
    float w = ea[e];
    int si = s * 256 + t;
    float v = ((float)hH[si] + (float)hL[si]) * w;
    unsafeAtomicAdd(&agg[d * 256 + t], v);      // hardware global_atomic_add_f32
}

// ---- layer GEMM: relu([agg|h](N x 512) @ wcatT^T + bias) -> bf16 hi/lo splits ----
// one wave per 16(M) x 32(N) C tile; A fragments reused across both N tiles.
__global__ void gnn_gemm_layer(const __bf16* __restrict__ aggH, const __bf16* __restrict__ aggL,
                               const __bf16* __restrict__ hH,  const __bf16* __restrict__ hL,
                               const __bf16* __restrict__ wTh, const __bf16* __restrict__ wTl,
                               const float* __restrict__ bias,
                               __bf16* __restrict__ outH, __bf16* __restrict__ outL) {
    const int lane  = threadIdx.x;
    const int ml    = lane & 15;
    const int half  = lane >> 4;
    const int mtile = blockIdx.x * 16;
    const int nt0   = blockIdx.y * 32;

    const long arow  = (long)(mtile + ml) * 256;
    const long brow0 = (long)(nt0 + ml) * 512;
    const long brow1 = (long)(nt0 + 16 + ml) * 512;

    v8f c0 = {0.f,0.f,0.f,0.f,0.f,0.f,0.f,0.f};
    v8f c1 = {0.f,0.f,0.f,0.f,0.f,0.f,0.f,0.f};

    // ---- first K half: A = agg split, B k-offset 0..255 ----
    for (int kt = 0; kt < 8; ++kt) {
        const int ka = (kt << 5) + (half << 3);
        const int kb = (kt << 5) + (half << 3);
        v16bf aH  = ldA(aggH + arow + ka);
        v16bf aL  = ldA(aggL + arow + ka);
        v16bf bH0 = ldB(wTh + brow0 + kb);
        v16bf bL0 = ldB(wTl + brow0 + kb);
        v16bf bH1 = ldB(wTh + brow1 + kb);
        v16bf bL1 = ldB(wTl + brow1 + kb);
        c0 = wmma_bf16(aH, bH0, c0); c0 = wmma_bf16(aH, bL0, c0); c0 = wmma_bf16(aL, bH0, c0);
        c1 = wmma_bf16(aH, bH1, c1); c1 = wmma_bf16(aH, bL1, c1); c1 = wmma_bf16(aL, bH1, c1);
    }
    // ---- second K half: A = h split, B k-offset 256..511 ----
    for (int kt = 0; kt < 8; ++kt) {
        const int ka = (kt << 5) + (half << 3);
        const int kb = 256 + (kt << 5) + (half << 3);
        v16bf aH  = ldA(hH + arow + ka);
        v16bf aL  = ldA(hL + arow + ka);
        v16bf bH0 = ldB(wTh + brow0 + kb);
        v16bf bL0 = ldB(wTl + brow0 + kb);
        v16bf bH1 = ldB(wTh + brow1 + kb);
        v16bf bL1 = ldB(wTl + brow1 + kb);
        c0 = wmma_bf16(aH, bH0, c0); c0 = wmma_bf16(aH, bL0, c0); c0 = wmma_bf16(aL, bH0, c0);
        c1 = wmma_bf16(aH, bH1, c1); c1 = wmma_bf16(aH, bL1, c1); c1 = wmma_bf16(aL, bH1, c1);
    }

    const float bn0  = bias[nt0 + ml];
    const float bn1  = bias[nt0 + 16 + ml];
    const int  mbase = mtile + half * 8;
    for (int v = 0; v < 8; ++v) {
        long rbase = (long)(mbase + v) * 256;
        float r0 = fmaxf(c0[v] + bn0, 0.0f);
        float r1 = fmaxf(c1[v] + bn1, 0.0f);
        __bf16 h0, l0, h1, l1;
        split_bf16(r0, h0, l0);
        split_bf16(r1, h1, l1);
        outH[rbase + nt0 + ml]      = h0;
        outL[rbase + nt0 + ml]      = l0;
        outH[rbase + nt0 + 16 + ml] = h1;
        outL[rbase + nt0 + 16 + ml] = l1;
    }
}

// ---- output GEMM: relu(h(N x 256) @ w_out + b_out) -> fp32 [N,128]; 16x32 C per wave ----
__global__ void gnn_gemm_out(const __bf16* __restrict__ hH, const __bf16* __restrict__ hL,
                             const __bf16* __restrict__ wTh, const __bf16* __restrict__ wTl,
                             const float* __restrict__ bias, float* __restrict__ out) {
    const int lane  = threadIdx.x;
    const int ml    = lane & 15;
    const int half  = lane >> 4;
    const int mtile = blockIdx.x * 16;
    const int nt0   = blockIdx.y * 32;

    const long arow  = (long)(mtile + ml) * 256;
    const long brow0 = (long)(nt0 + ml) * 256;
    const long brow1 = (long)(nt0 + 16 + ml) * 256;

    v8f c0 = {0.f,0.f,0.f,0.f,0.f,0.f,0.f,0.f};
    v8f c1 = {0.f,0.f,0.f,0.f,0.f,0.f,0.f,0.f};

    for (int kt = 0; kt < 8; ++kt) {
        const int ka = (kt << 5) + (half << 3);
        const int kb = (kt << 5) + (half << 3);
        v16bf aH  = ldA(hH + arow + ka);
        v16bf aL  = ldA(hL + arow + ka);
        v16bf bH0 = ldB(wTh + brow0 + kb);
        v16bf bL0 = ldB(wTl + brow0 + kb);
        v16bf bH1 = ldB(wTh + brow1 + kb);
        v16bf bL1 = ldB(wTl + brow1 + kb);
        c0 = wmma_bf16(aH, bH0, c0); c0 = wmma_bf16(aH, bL0, c0); c0 = wmma_bf16(aL, bH0, c0);
        c1 = wmma_bf16(aH, bH1, c1); c1 = wmma_bf16(aH, bL1, c1); c1 = wmma_bf16(aL, bH1, c1);
    }

    const float bn0  = bias[nt0 + ml];
    const float bn1  = bias[nt0 + 16 + ml];
    const int  mbase = mtile + half * 8;
    for (int v = 0; v < 8; ++v) {
        long rbase = (long)(mbase + v) * 128;
        out[rbase + nt0 + ml]      = fmaxf(c0[v] + bn0, 0.0f);
        out[rbase + nt0 + 16 + ml] = fmaxf(c1[v] + bn1, 0.0f);
    }
}

extern "C" void kernel_launch(void* const* d_in, const int* in_sizes, int n_in,
                              void* d_out, int out_size, void* d_ws, size_t ws_size,
                              hipStream_t stream) {
    const float* x      = (const float*)d_in[0];
    const int*   ei     = (const int*)d_in[1];
    const float* ea     = (const float*)d_in[2];
    const float* w_rel  = (const float*)d_in[3];
    const float* b_rel  = (const float*)d_in[4];
    const float* w_root = (const float*)d_in[5];
    const float* w_out  = (const float*)d_in[6];
    const float* b_out  = (const float*)d_in[7];

    const int N  = in_sizes[0] / 256;   // 20000
    const int E  = in_sizes[2];         // 320000
    const int NH = N * 256;

    // ---- workspace carve-up ----
    char* ws = (char*)d_ws;
    size_t off = 0;
    auto carve = [&](size_t bytes) -> char* {
        char* p = ws + off;
        off += (bytes + 255) & ~(size_t)255;
        return p;
    };
    __bf16* hH[2]  = { (__bf16*)carve((size_t)NH * 2), (__bf16*)carve((size_t)NH * 2) };
    __bf16* hL[2]  = { (__bf16*)carve((size_t)NH * 2), (__bf16*)carve((size_t)NH * 2) };
    float*  agg    = (float*)carve((size_t)NH * 4);
    __bf16* aggH   = (__bf16*)carve((size_t)NH * 2);
    __bf16* aggL   = (__bf16*)carve((size_t)NH * 2);
    __bf16* wcatH  = (__bf16*)carve((size_t)3 * 256 * 512 * 2);
    __bf16* wcatL  = (__bf16*)carve((size_t)3 * 256 * 512 * 2);
    __bf16* woutH  = (__bf16*)carve((size_t)128 * 256 * 2);
    __bf16* woutL  = (__bf16*)carve((size_t)128 * 256 * 2);
    (void)ws_size; (void)n_in; (void)out_size;

    const int cvtBlocks = (NH + 255) / 256;

    // one-time prep (deterministic, redone every call)
    gnn_cvt_split<<<cvtBlocks, 256, 0, stream>>>(x, hH[0], hL[0], NH);
    gnn_prep_wcat<<<1536, 256, 0, stream>>>(w_rel, w_root, wcatH, wcatL);
    gnn_prep_wout<<<128, 256, 0, stream>>>(w_out, woutH, woutL);

    int cur = 0;
    for (int l = 0; l < 3; ++l) {
        hipMemsetAsync(agg, 0, (size_t)NH * 4, stream);
        gnn_scatter<<<E, 256, 0, stream>>>(ei, ea, hH[cur], hL[cur], agg, E);
        gnn_cvt_split<<<cvtBlocks, 256, 0, stream>>>(agg, aggH, aggL, NH);
        gnn_gemm_layer<<<dim3(N / 16, 8), 32, 0, stream>>>(
            aggH, aggL, hH[cur], hL[cur],
            wcatH + (size_t)l * 256 * 512, wcatL + (size_t)l * 256 * 512,
            b_rel + l * 256,
            hH[1 - cur], hL[1 - cur]);
        cur = 1 - cur;
    }

    gnn_gemm_out<<<dim3(N / 16, 4), 32, 0, stream>>>(
        hH[cur], hL[cur], woutH, woutL, b_out, (float*)d_out);
}
